// LHCOptimizedSNN_42210938585482
// MI455X (gfx1250) — matrix operand order repaired and checked by reading, not compiled, not run on hardware
//
#include <hip/hip_runtime.h>
#include <hip/hip_bf16.h>

typedef __bf16 bf16_t;
typedef __attribute__((ext_vector_type(16))) __bf16 v16bf;
typedef __attribute__((ext_vector_type(8)))  __bf16 v8bf;
typedef __attribute__((ext_vector_type(8)))  float  v8f;
typedef __attribute__((ext_vector_type(4)))  float  v4f;

#define BATCH   16384
#define THRESH  1.2f

static __device__ __forceinline__ bf16_t f2bf(float f) {
  __hip_bfloat16 t = __float2bfloat16(f);
  return __builtin_bit_cast(bf16_t, t);
}

// ---------------------------------------------------------------------------
// Zero the 6 float stat accumulators (ssum0,any0,ssum1,any1,ssum2,any2)
// ---------------------------------------------------------------------------
__global__ void zero_stats_kernel(float* __restrict__ s) {
  if (threadIdx.x < 6) s[threadIdx.x] = 0.0f;
}

// ---------------------------------------------------------------------------
// Pre-swizzle weights fp32[K,srcN] (row major) -> bf16 B-operand layout:
//   dst[(((kb*NT + nb)*32 + lane)*16 + j)]
//   lane covers column n = nb*16 + (lane&15)
//   element j covers K = kb*32 + (lane>>4)*8 + (j&7) + (j>>3)*16      (ISA B 32x16 layout)
// Columns >= srcN are zero padding (used to pad W2 N=4 -> 16).
// ---------------------------------------------------------------------------
__global__ void convW_kernel(const float* __restrict__ W, bf16_t* __restrict__ dst,
                             int srcN, int NT, int total) {
  int idx = blockIdx.x * blockDim.x + threadIdx.x;
  if (idx >= total) return;
  int j    = idx & 15;
  int lane = (idx >> 4) & 31;
  int blk  = idx >> 9;            // kb*NT + nb
  int nb   = blk % NT;
  int kb   = blk / NT;
  int k = kb * 32 + (lane >> 4) * 8 + (j & 7) + (j >> 3) * 16;
  int n = nb * 16 + (lane & 15);
  float v = (n < srcN) ? W[(size_t)k * srcN + n] : 0.0f;
  dst[idx] = f2bf(v);
}

// ---------------------------------------------------------------------------
// Fused GEMM (bf16 WMMA, f32 accumulate) + 5-step LIF dynamics + stats.
//   A: [M,K] fp32 (layer 0, converted in-register) or bf16 (later layers)
//   Bw: pre-swizzled bf16 weights (see convW_kernel)
//   One wave owns a 16-row M tile and NTW 16-col N tiles (full N => A read once).
//   128-thread blocks (4 waves) to spread queues across many WGPs.
// ---------------------------------------------------------------------------
template<int NTW, bool A_F32, bool STORE_F32>
__global__ __launch_bounds__(128) void lif_gemm_kernel(
    const void* __restrict__ A, const bf16_t* __restrict__ Bw,
    const float* __restrict__ bias, void* __restrict__ H,
    float* __restrict__ stats, int K, int ldH, int nvalid, int ntTotal) {

  const int lane  = threadIdx.x & 31;
  const int wave  = threadIdx.x >> 5;
  const int nwave = blockDim.x >> 5;
  const int mtile = blockIdx.x * nwave + wave;
  const int row0  = mtile * 16;
  const int arow  = row0 + (lane & 15);
  const int kbase = (lane >> 4) * 8;     // A operand per-lane K base
  const int nbg0  = blockIdx.y * NTW;    // first N tile of this wave

  v8f acc[NTW];
  const v8f vzero = {0.f, 0.f, 0.f, 0.f, 0.f, 0.f, 0.f, 0.f};
#pragma unroll
  for (int i = 0; i < NTW; ++i) acc[i] = vzero;

  const v16bf* bbase = (const v16bf*)Bw;
  const int KB = K >> 5;

  for (int kb = 0; kb < KB; ++kb) {
    const int k0 = kb * 32 + kbase;
    v16bf a;
    if (A_F32) {
      // ISA A 16x32 bf16 layout: lanes 0-15 hold K {0..7,16..23}, lanes 16-31 {8..15,24..31}
      const float* ap = (const float*)A + (size_t)arow * K + k0;
      v4f x0 = __builtin_nontemporal_load((const v4f*)(ap));
      v4f x1 = __builtin_nontemporal_load((const v4f*)(ap + 4));
      v4f x2 = __builtin_nontemporal_load((const v4f*)(ap + 16));
      v4f x3 = __builtin_nontemporal_load((const v4f*)(ap + 20));
#pragma unroll
      for (int j = 0; j < 4; ++j) {
        a[j]      = f2bf(x0[j]);
        a[4 + j]  = f2bf(x1[j]);
        a[8 + j]  = f2bf(x2[j]);
        a[12 + j] = f2bf(x3[j]);
      }
    } else {
      const bf16_t* ap = (const bf16_t*)A + (size_t)arow * K + k0;
      v8bf lo = *(const v8bf*)(ap);
      v8bf hi = *(const v8bf*)(ap + 16);
#pragma unroll
      for (int j = 0; j < 8; ++j) { a[j] = lo[j]; a[8 + j] = hi[j]; }
    }

    const v16bf* bp = bbase + ((size_t)kb * ntTotal + nbg0) * 32 + lane;
#pragma unroll
    for (int nb = 0; nb < NTW; ++nb) {
      v16bf b = bp[nb * 32];
      acc[nb] = __builtin_amdgcn_wmma_f32_16x16x32_bf16(
          false, a, false, b, (short)0, acc[nb], false, false);
    }
  }

  // ---- 5-step LIF on the accumulators (C layout: VGPR r -> M=r / r+8) ----
  float ssum = 0.0f, anyc = 0.0f;
  const int colb = lane & 15;
  const int mro  = row0 + (lane >> 4) * 8;

#pragma unroll
  for (int nb = 0; nb < NTW; ++nb) {
    const int col = (nbg0 + nb) * 16 + colb;
    const float bv = (col < nvalid) ? bias[col] : 0.0f;
#pragma unroll
    for (int r = 0; r < 8; ++r) {
      float cur = acc[nb][r] + bv;
      float v = 0.0f, last = 0.0f, esum = 0.0f, any = 0.0f;
#pragma unroll
      for (int t = 0; t < 5; ++t) {
        v = 0.9f * v + 0.1f * cur;
        float sp = (v >= THRESH) ? 2.0f : 0.0f;  // forward spike value is 2*hard
        v -= sp * THRESH;
        esum += sp;
        any = fmaxf(any, sp);
        last = sp;
      }
      if (col < nvalid) {
        ssum += esum;
        anyc += (any > 0.0f) ? 1.0f : 0.0f;
      }
      const int row = mro + r;
      if (STORE_F32) {
        if (col < nvalid) ((float*)H)[(size_t)row * nvalid + col] = last;
      } else {
        ((bf16_t*)H)[(size_t)row * ldH + col] = f2bf(last);
      }
    }
  }

  // ---- block reduction of stats, one atomicAdd pair per block ----
#pragma unroll
  for (int off = 16; off > 0; off >>= 1) {
    ssum += __shfl_xor(ssum, off, 32);
    anyc += __shfl_xor(anyc, off, 32);
  }
  __shared__ float red[16];
  if (lane == 0) { red[wave * 2] = ssum; red[wave * 2 + 1] = anyc; }
  __syncthreads();
  if (threadIdx.x == 0) {
    float s = 0.0f, a = 0.0f;
    for (int w = 0; w < nwave; ++w) { s += red[2 * w]; a += red[2 * w + 1]; }
    atomicAdd(&stats[0], s);
    atomicAdd(&stats[1], a);
  }
}

// ---------------------------------------------------------------------------
// Output layer: logits = (1 - 0.9^5) * (h2 @ Wout + bout)
// ---------------------------------------------------------------------------
__global__ void out_kernel(const float* __restrict__ h2,
                           const float* __restrict__ Wout,
                           const float* __restrict__ bout,
                           float* __restrict__ out, int M) {
  int r = blockIdx.x * blockDim.x + threadIdx.x;
  if (r >= M) return;
  float h0 = h2[(size_t)r * 4 + 0];
  float h1 = h2[(size_t)r * 4 + 1];
  float h2v = h2[(size_t)r * 4 + 2];
  float h3 = h2[(size_t)r * 4 + 3];
  const float f = 0.40951f;  // 1 - 0.9^5
#pragma unroll
  for (int c = 0; c < 2; ++c) {
    float s = bout[c] + h0 * Wout[0 * 2 + c] + h1 * Wout[1 * 2 + c]
                      + h2v * Wout[2 * 2 + c] + h3 * Wout[3 * 2 + c];
    out[(size_t)r * 2 + c] = f * s;
  }
}

__global__ void finalize_kernel(const float* __restrict__ stats,
                                float* __restrict__ out, int off) {
  if (threadIdx.x == 0 && blockIdx.x == 0) {
    const float B = (float)BATCH;
    float sr0 = stats[0] / (5.0f * B * 256.0f);
    float sr1 = stats[2] / (5.0f * B * 128.0f);
    float sr2 = stats[4] / (5.0f * B * 4.0f);
    float ar0 = stats[1] / (B * 256.0f);
    float ar1 = stats[3] / (B * 128.0f);
    float ar2 = stats[5] / (B * 4.0f);
    out[off]     = (sr0 + sr1 + sr2) * (1.0f / 3.0f);
    out[off + 1] = (ar0 + ar1 + ar2) * (1.0f / 3.0f);
  }
}

// ---------------------------------------------------------------------------
extern "C" void kernel_launch(void* const* d_in, const int* in_sizes, int n_in,
                              void* d_out, int out_size, void* d_ws, size_t ws_size,
                              hipStream_t stream) {
  const float* x    = (const float*)d_in[0];
  const float* W0   = (const float*)d_in[1];
  const float* b0   = (const float*)d_in[2];
  const float* W1   = (const float*)d_in[3];
  const float* b1   = (const float*)d_in[4];
  const float* W2   = (const float*)d_in[5];
  const float* b2   = (const float*)d_in[6];
  const float* Wout = (const float*)d_in[7];
  const float* bout = (const float*)d_in[8];

  char* ws = (char*)d_ws;
  // Workspace layout (256B aligned segments)
  float*  stats = (float*)ws;                                   // 6 floats
  bf16_t* W0b   = (bf16_t*)(ws + 256);                          // 4096x256  -> 2 MB
  bf16_t* W1b   = (bf16_t*)(ws + 256 + 2097152);                // 256x128   -> 64 KB
  bf16_t* W2b   = (bf16_t*)(ws + 256 + 2097152 + 65536);        // 128x16pad -> 4 KB
  bf16_t* h0    = (bf16_t*)(ws + 256 + 2097152 + 65536 + 4096);           // 16384x256 bf16 = 8 MB
  bf16_t* h1    = (bf16_t*)(ws + 256 + 2097152 + 65536 + 4096 + 8388608); // 16384x128 bf16 = 4 MB
  float*  h2    = (float*)(ws + 256 + 2097152 + 65536 + 4096 + 8388608 + 4194304); // 16384x4 f32

  zero_stats_kernel<<<1, 32, 0, stream>>>(stats);

  // W0: K=4096, N=256 (NT=16): total = 128*16*512 = 1,048,576 elems
  convW_kernel<<<(1048576 + 255) / 256, 256, 0, stream>>>(W0, W0b, 256, 16, 1048576);
  // W1: K=256, N=128 (NT=8): total = 8*8*512 = 32,768
  convW_kernel<<<(32768 + 255) / 256, 256, 0, stream>>>(W1, W1b, 128, 8, 32768);
  // W2: K=128, N=4 padded to 16 (NT=1): total = 4*1*512 = 2,048
  convW_kernel<<<(2048 + 255) / 256, 256, 0, stream>>>(W2, W2b, 4, 1, 2048);

  // Layer 0: x[16384,4096] fp32 -> h0[16384,256] bf16.
  // Single N group (16 tiles/wave) => x is streamed from HBM exactly once.
  // 4 waves/block, 256 blocks.
  lif_gemm_kernel<16, true, false><<<dim3(BATCH / 64, 1), 128, 0, stream>>>(
      (const void*)x, W0b, b0, (void*)h0, stats + 0, 4096, 256, 256, 16);

  // Layer 1: h0[16384,256] bf16 -> h1[16384,128] bf16.  8 N-tiles per wave.
  lif_gemm_kernel<8, false, false><<<dim3(BATCH / 64, 1), 128, 0, stream>>>(
      (const void*)h0, W1b, b1, (void*)h1, stats + 2, 256, 128, 128, 8);

  // Layer 2: h1[16384,128] bf16 -> h2[16384,4] fp32 (N padded to 16, 4 valid cols).
  lif_gemm_kernel<1, false, true><<<dim3(BATCH / 64, 1), 128, 0, stream>>>(
      (const void*)h1, W2b, b2, (void*)h2, stats + 4, 128, 4, 4, 1);

  // Output layer + scalars.
  out_kernel<<<(BATCH + 255) / 256, 256, 0, stream>>>(h2, Wout, bout, (float*)d_out, BATCH);
  finalize_kernel<<<1, 1, 0, stream>>>(stats, (float*)d_out, BATCH * 2);
}
